// XTAttention_39135742001592
// MI455X (gfx1250) — compile-verified, matrix operand order checked by reading, not run on hardware
//
#include <hip/hip_runtime.h>
#include <hip/hip_bf16.h>

// MI455X / gfx1250, wave32. All matmuls via v_wmma_f32_16x16x32_bf16.
// Double-buffered LDS staging; async global->LDS (ASYNCcnt) for bf16 tiles.

typedef __bf16 bf16_t;
typedef __attribute__((ext_vector_type(16))) __bf16 v16bf;
typedef __attribute__((ext_vector_type(8)))  __bf16 v8bf;
typedef __attribute__((ext_vector_type(4)))  __bf16 v4bf;
typedef __attribute__((ext_vector_type(8)))  float v8f;
typedef __attribute__((ext_vector_type(4)))  float v4f;
typedef __attribute__((ext_vector_type(2)))  float v2f;

#define D_DIM 1024
#define XLS_STRIDE 40            // 32 + pad (16B-aligned rows, bank spread)
#define PLS_STRIDE 136           // 128 + pad
#define XBUF (128 * XLS_STRIDE)  // one staging buffer (elements)

// Load a 16x32 bf16 WMMA operand fragment from LDS.
// A-layout (M x K): lane l -> row (l&15), K groups {(l>>4)*8..+7} and {16+(l>>4)*8..+7}.
// Serves B too when the LDS image is stored transposed [N][K].
__device__ __forceinline__ v16bf frag_ld(const bf16_t* base, int row0, int kbase,
                                         int stride, int lane) {
  const int half = lane >> 4;
  const bf16_t* p = base + (row0 + (lane & 15)) * stride + kbase + half * 8;
  v8bf lo = *(const v8bf*)(p);
  v8bf hi = *(const v8bf*)(p + 16);
  return __builtin_shufflevector(lo, hi, 0,1,2,3,4,5,6,7,8,9,10,11,12,13,14,15);
}

// ---- register staging helpers (f32 global -> regs -> bf16 LDS) ----
__device__ __forceinline__ void stage_x(const float* __restrict__ Xg, int k0,
                                        int xrow, int xkq, v4f xr[4]) {
  #pragma unroll
  for (int it = 0; it < 4; ++it)
    xr[it] = *(const v4f*)(Xg + (size_t)(xrow + it * 32) * D_DIM + k0 + xkq);
}
__device__ __forceinline__ void commit_x(bf16_t* Xd, int xrow, int xkq, const v4f xr[4]) {
  #pragma unroll
  for (int it = 0; it < 4; ++it) {
    v4bf o = { (bf16_t)xr[it].x, (bf16_t)xr[it].y, (bf16_t)xr[it].z, (bf16_t)xr[it].w };
    *(v4bf*)(Xd + (xrow + it * 32) * XLS_STRIDE + xkq) = o;
  }
}
__device__ __forceinline__ void stage_w(const float* __restrict__ Wg, int k0, int col0,
                                        int wn, int wk, v2f wr[8]) {
  #pragma unroll
  for (int it = 0; it < 8; ++it)
    wr[it] = *(const v2f*)(Wg + (size_t)(k0 + wk + it * 4) * D_DIM + col0 + wn);
}
__device__ __forceinline__ void commit_w(bf16_t* Wd, int wn, int wk, const v2f wr[8]) {
  #pragma unroll
  for (int it = 0; it < 8; ++it) {           // transposed: Wd[n][k]
    Wd[wn * XLS_STRIDE + wk + it * 4]       = (bf16_t)wr[it].x;
    Wd[(wn + 1) * XLS_STRIDE + wk + it * 4] = (bf16_t)wr[it].y;
  }
}

// Y(128x128) = X(128x1024) @ W[:, col0:col0+128], bf16 result into LDS.
// Double-buffered: one barrier per K-chunk; next chunk staged in regs during WMMA.
// Safe: LDS writes at iter kc hit the buffer last read at compute(kc-2), and
// arrival at barrier(kc-1) implies compute(kc-2) finished.
__device__ __forceinline__ void proj_gemm(
    const float* __restrict__ Xg, const float* __restrict__ Wg, int col0,
    bf16_t* Xls, bf16_t* Wls, bf16_t* dst, int transposeDst, int tid)
{
  const int lane = tid & 31, wave = tid >> 5;
  const int xrow = tid >> 3, xkq = (tid & 7) * 4;
  const int wn = (tid & 63) * 2, wk = tid >> 6;
  v8f acc[8] = {};
  v4f xr[4]; v2f wr[8];
  stage_x(Xg, 0, xrow, xkq, xr);
  stage_w(Wg, 0, col0, wn, wk, wr);
  for (int kc = 0; kc < 32; ++kc) {
    bf16_t* Xd = Xls + (kc & 1) * XBUF;
    bf16_t* Wd = Wls + (kc & 1) * XBUF;
    commit_x(Xd, xrow, xkq, xr);
    commit_w(Wd, wn, wk, wr);
    if (kc + 1 < 32) {
      stage_x(Xg, (kc + 1) * 32, xrow, xkq, xr);      // overlap with WMMA below
      stage_w(Wg, (kc + 1) * 32, col0, wn, wk, wr);
    }
    __syncthreads();
    v16bf a = frag_ld(Xd, wave * 16, 0, XLS_STRIDE, lane);
    #pragma unroll
    for (int c = 0; c < 8; ++c) {
      v16bf b = frag_ld(Wd, c * 16, 0, XLS_STRIDE, lane);
      acc[c] = __builtin_amdgcn_wmma_f32_16x16x32_bf16(
          false, a, false, b, (short)0, acc[c], false, false);
    }
  }
  // C/D layout: row = r + 8*(lane>>4), col = lane&15 (per 16x16 tile)
  const int nb = lane & 15, half = lane >> 4;
  if (transposeDst) {
    #pragma unroll
    for (int c = 0; c < 8; ++c) {             // 8 consecutive m per column -> b128 store
      v8bf o = { (bf16_t)acc[c][0], (bf16_t)acc[c][1], (bf16_t)acc[c][2], (bf16_t)acc[c][3],
                 (bf16_t)acc[c][4], (bf16_t)acc[c][5], (bf16_t)acc[c][6], (bf16_t)acc[c][7] };
      *(v8bf*)(dst + (c * 16 + nb) * PLS_STRIDE + wave * 16 + half * 8) = o;
    }
  } else {
    #pragma unroll
    for (int c = 0; c < 8; ++c)
      #pragma unroll
      for (int r = 0; r < 8; ++r)
        dst[(wave * 16 + r + half * 8) * PLS_STRIDE + c * 16 + nb] = (bf16_t)acc[c][r];
  }
}

// One workgroup per (b,h): 7 projections + 3-stream scores + softmax + P@V.
__global__ __launch_bounds__(256, 1) void attn_fused_kernel(
    const float* __restrict__ words, const float* __restrict__ position,
    const float* __restrict__ conscious,
    const float* __restrict__ Wq_w, const float* __restrict__ Wk_w,
    const float* __restrict__ Wv_w,
    const float* __restrict__ Wq_p, const float* __restrict__ Wk_p,
    const float* __restrict__ Wq_c, const float* __restrict__ Wk_c,
    bf16_t* __restrict__ Ows)
{
  __shared__ __align__(16) bf16_t Xls[2 * XBUF];
  __shared__ __align__(16) bf16_t Wls[2 * XBUF];
  __shared__ __align__(16) bf16_t bufA[128 * PLS_STRIDE];  // Q_s, later P
  __shared__ __align__(16) bf16_t bufB[128 * PLS_STRIDE];  // K_s^T, later V^T

  const int tid = threadIdx.x, lane = tid & 31, wave = tid >> 5;
  const int b = blockIdx.x >> 3, h = blockIdx.x & 7;
  const int col0 = h * 128;
  const float* Xs[3] = { words + (size_t)b * 128 * D_DIM,
                         position + (size_t)b * 128 * D_DIM,
                         conscious + (size_t)b * 128 * D_DIM };
  const float* WQ[3] = { Wq_w, Wq_p, Wq_c };
  const float* WK[3] = { Wk_w, Wk_p, Wk_c };

  v8f accS[8] = {};
  for (int s = 0; s < 3; ++s) {
    proj_gemm(Xs[s], WQ[s], col0, Xls, Wls, bufA, 0, tid);  // Q_s [i][j]
    proj_gemm(Xs[s], WK[s], col0, Xls, Wls, bufB, 1, tid);  // K_s^T [d][j]
    __syncthreads();
    #pragma unroll
    for (int kk = 0; kk < 4; ++kk) {           // S += Q_s @ K_s (j = 128)
      v16bf a = frag_ld(bufA, wave * 16, kk * 32, PLS_STRIDE, lane);
      #pragma unroll
      for (int c = 0; c < 8; ++c) {
        v16bf bb = frag_ld(bufB, c * 16, kk * 32, PLS_STRIDE, lane);
        accS[c] = __builtin_amdgcn_wmma_f32_16x16x32_bf16(
            false, a, false, bb, (short)0, accS[c], false, false);
      }
    }
  }
  __syncthreads();  // all S-operand reads of bufA/bufB complete

  // Row softmax entirely in registers: row m = wave*16 + r + 8*half spans
  // 8 tiles x 16 lanes of one half-wave -> 4 shfl_xor reductions.
  const float SCALE = 0.08838834764831845f;  // 128^-0.5
  const int nb = lane & 15, half = lane >> 4;
  #pragma unroll
  for (int r = 0; r < 8; ++r) {
    float v[8];
    float mx = -3.402823466e38f;
    #pragma unroll
    for (int c = 0; c < 8; ++c) { v[c] = accS[c][r] * SCALE; mx = fmaxf(mx, v[c]); }
    #pragma unroll
    for (int off = 1; off < 16; off <<= 1) mx = fmaxf(mx, __shfl_xor(mx, off, 32));
    float sum = 0.f;
    #pragma unroll
    for (int c = 0; c < 8; ++c) { v[c] = __expf(v[c] - mx); sum += v[c]; }
    #pragma unroll
    for (int off = 1; off < 16; off <<= 1) sum += __shfl_xor(sum, off, 32);
    const float inv = 1.f / sum;
    const int m = wave * 16 + r + half * 8;
    #pragma unroll
    for (int c = 0; c < 8; ++c)
      bufA[m * PLS_STRIDE + c * 16 + nb] = (bf16_t)(v[c] * inv);  // P [i][j]
  }

  proj_gemm(Xs[0], Wv_w, col0, Xls, Wls, bufB, 1, tid);  // V^T [d][j]
  __syncthreads();

  v8f accO[8] = {};
  #pragma unroll
  for (int kk = 0; kk < 4; ++kk) {             // O = P @ V
    v16bf a = frag_ld(bufA, wave * 16, kk * 32, PLS_STRIDE, lane);
    #pragma unroll
    for (int c = 0; c < 8; ++c) {
      v16bf bb = frag_ld(bufB, c * 16, kk * 32, PLS_STRIDE, lane);
      accO[c] = __builtin_amdgcn_wmma_f32_16x16x32_bf16(
          false, a, false, bb, (short)0, accO[c], false, false);
    }
  }
  #pragma unroll
  for (int c = 0; c < 8; ++c)
    #pragma unroll
    for (int r = 0; r < 8; ++r) {
      const int m = wave * 16 + r + half * 8;
      Ows[((size_t)b * 128 + m) * D_DIM + col0 + c * 16 + nb] = (bf16_t)accO[c][r];
    }
}

// out = O(bf16) @ Wo + bo, f32 out. A-tiles fetched with async global->LDS.
// Race-free async double-buffer: drain own ASYNCcnt BEFORE the barrier, issue
// the next buffer's transfers AFTER the barrier (every wave's compute(kc-1) --
// the last reader of that buffer -- precedes its arrival at barrier(kc)).
__global__ __launch_bounds__(256, 1) void out_proj_kernel(
    const bf16_t* __restrict__ Ows, const float* __restrict__ Wo,
    const float* __restrict__ bo, float* __restrict__ out)
{
  __shared__ __align__(16) bf16_t Als[2 * XBUF];
  __shared__ __align__(16) bf16_t Bls[2 * XBUF];
  const int tid = threadIdx.x, lane = tid & 31, wave = tid >> 5;
  const int m0 = blockIdx.x * 128;
  const int n0 = blockIdx.y * 128;
  const int arow = tid >> 2, akq = (tid & 3) * 8;
  const int wn = (tid & 63) * 2, wk = tid >> 6;

  auto issue_async = [&](int kc) {
    bf16_t* dstbuf = Als + (kc & 1) * XBUF;
    const int k0 = kc * 32;
    #pragma unroll
    for (int it = 0; it < 2; ++it) {
      const int r = arow + it * 64;
      const bf16_t* g = Ows + (size_t)(m0 + r) * D_DIM + k0 + akq;
      // low 32 bits of a flat shared pointer == wave-relative LDS byte address
      const unsigned ldsoff = (unsigned)(uintptr_t)(dstbuf + r * XLS_STRIDE + akq);
      asm volatile("global_load_async_to_lds_b128 %0, %1, off"
                   :: "v"(ldsoff), "v"(g) : "memory");
    }
  };

  v2f wr[8];
  stage_w(Wo, 0, n0, wn, wk, wr);
  issue_async(0);
  v8f acc[8] = {};
  for (int kc = 0; kc < 32; ++kc) {
    bf16_t* Ab = Als + (kc & 1) * XBUF;
    bf16_t* Bb = Bls + (kc & 1) * XBUF;
    commit_w(Bb, wn, wk, wr);                       // W chunk kc -> LDS (ds)
    if (kc + 1 < 32)
      stage_w(Wo, (kc + 1) * 32, n0, wn, wk, wr);   // next W chunk -> regs
    // Own buf-kc async transfers (issued a full compute-phase ago) must land
    // before we signal readiness; barrier then publishes everyone's data.
    asm volatile("s_wait_asynccnt 0x0" ::: "memory");
    __syncthreads();
    if (kc + 1 < 32)
      issue_async(kc + 1);   // after barrier: no wave still reads buf (kc+1)&1
    v16bf a = frag_ld(Ab, wave * 16, 0, XLS_STRIDE, lane);
    #pragma unroll
    for (int c = 0; c < 8; ++c) {
      v16bf bb = frag_ld(Bb, c * 16, 0, XLS_STRIDE, lane);
      acc[c] = __builtin_amdgcn_wmma_f32_16x16x32_bf16(
          false, a, false, bb, (short)0, acc[c], false, false);
    }
  }
  const int nb = lane & 15, half = lane >> 4;
  #pragma unroll
  for (int c = 0; c < 8; ++c) {
    const int n = n0 + c * 16 + nb;
    const float bias = bo[n];
    #pragma unroll
    for (int r = 0; r < 8; ++r) {
      const int m = m0 + wave * 16 + r + half * 8;
      out[(size_t)m * D_DIM + n] = acc[c][r] + bias;
    }
  }
}

extern "C" void kernel_launch(void* const* d_in, const int* in_sizes, int n_in,
                              void* d_out, int out_size, void* d_ws, size_t ws_size,
                              hipStream_t stream) {
  const float* words     = (const float*)d_in[0];
  const float* position  = (const float*)d_in[1];
  const float* conscious = (const float*)d_in[2];
  const float* Wq_w = (const float*)d_in[3];
  const float* Wk_w = (const float*)d_in[4];
  const float* Wv_w = (const float*)d_in[5];
  const float* Wq_p = (const float*)d_in[6];
  const float* Wk_p = (const float*)d_in[7];
  const float* Wq_c = (const float*)d_in[8];
  const float* Wk_c = (const float*)d_in[9];
  const float* Wo   = (const float*)d_in[10];
  const float* bo   = (const float*)d_in[11];
  float* out = (float*)d_out;
  bf16_t* Ows = (bf16_t*)d_ws;   // 16384*1024 bf16 = 32 MiB scratch

  attn_fused_kernel<<<dim3(128 * 8), dim3(256), 0, stream>>>(
      words, position, conscious, Wq_w, Wk_w, Wv_w, Wq_p, Wk_p, Wq_c, Wk_c, Ows);
  out_proj_kernel<<<dim3(128, 8), dim3(256), 0, stream>>>(Ows, Wo, bo, out);
}
